// add_ResnetBlock_77360950935559
// MI455X (gfx1250) — compile-verified, matrix operand order checked by reading, not run on hardware
//
#include <hip/hip_runtime.h>

typedef __attribute__((ext_vector_type(8)))  _Float16     v8h;
typedef __attribute__((ext_vector_type(16))) _Float16     v16h;
typedef __attribute__((ext_vector_type(8)))  float        v8f;
typedef __attribute__((ext_vector_type(4)))  unsigned int u32x4;

// ---------------------------------------------------------------------------
// Problem constants (from reference): x [4,64,40,40] f32, w [64,64,3,3] f32.
// K = 64*9 = 576 = 18 chunks of 32 (one WMMA k-slice each).
// ---------------------------------------------------------------------------
#define NB   4
#define CC   64
#define HH   40
#define WW   40
#define KK   576
#define NCHK 18

// Convert fp32 weights (already laid out [Cout][Cin*9] when flattened) to f16.
__global__ void cvt_weights(const float* __restrict__ w,
                            _Float16* __restrict__ wA, int n) {
  int i = blockIdx.x * 256 + threadIdx.x;
  if (i < n) wA[i] = (_Float16)w[i];
}

// Packed |p - w| for one lane's 16 A-matrix elements (two 8-f16 k-runs).
static __device__ __forceinline__ v16h mkAbsA(v8h p0, v8h p1, v8h w0, v8h w1) {
  const v8h d0 = p0 - w0;                 // v_pk_add_f16 neg_lo/neg_hi
  const v8h d1 = p1 - w1;
  const u32x4 b0 = __builtin_bit_cast(u32x4, d0) & 0x7FFF7FFFu;  // packed |.|
  const u32x4 b1 = __builtin_bit_cast(u32x4, d1) & 0x7FFF7FFFu;
  return __builtin_shufflevector(
      __builtin_bit_cast(v8h, b0), __builtin_bit_cast(v8h, b1),
      0, 1, 2, 3, 4, 5, 6, 7, 8, 9, 10, 11, 12, 13, 14, 15);
}

// One AdderNet layer: out[n,co,row,col] = f(-sum_k |p[k] - w[co,k]|)
//   IN_HALF : input activations are f16 (layer 2 reads layer 1's f16 scratch)
//   FINAL   : apply 0.1*relu(.) + x residual and write f32, else relu -> f16
template <bool IN_HALF, bool FINAL>
__global__ __launch_bounds__(256)
void adder_layer(const void* __restrict__ inRaw,
                 const float* __restrict__ xres,
                 const _Float16* __restrict__ wA,
                 void* __restrict__ outRaw) {
  // Patch matrix for this output row: P[col][k], k = ci*9 + kh*3 + kw.
  // Row stride 576 f16 = 1152 B (16B aligned); chunk offsets are 16B aligned.
  __shared__ _Float16 P[WW * KK];

  const int n   = blockIdx.x / HH;
  const int row = blockIdx.x % HH;
  const int tid = threadIdx.x;

  // ---------------- build P (all 256 threads) ----------------
  const float*    inF = (const float*)inRaw;
  const _Float16* inH = (const _Float16*)inRaw;
  for (int idx = tid; idx < WW * CC; idx += 256) {
    const int col = idx >> 6;          // 0..39
    const int ci  = idx & 63;          // 0..63
    _Float16* dst = &P[col * KK + ci * 9];
#pragma unroll
    for (int kh = 0; kh < 3; ++kh) {
      const int r = row + kh - 1;
#pragma unroll
      for (int kw = 0; kw < 3; ++kw) {
        const int cc = col + kw - 1;
        _Float16 v = (_Float16)0.f;
        if (r >= 0 && r < HH && cc >= 0 && cc < WW) {
          const long off = (((long)(n * CC + ci)) * HH + r) * WW + cc;
          v = IN_HALF ? inH[off] : (_Float16)inF[off];
        }
        dst[kh * 3 + kw] = v;
      }
    }
  }
  __syncthreads();

  // ---------------- per-wave WMMA-reduced SAD ----------------
  const int wid  = tid >> 5;
  const int lane = tid & 31;
  const int cobase = (wid & 3) * 16;       // 4 co-groups of 16
  const int col00  = (wid >> 2) * 20;      // 2 column halves of 20
  const int mrow   = lane & 15;            // A-matrix row this lane owns
  const int klo    = (lane < 16) ? 0 : 8;  // ISA 16-bit A layout k split

  // B = all-ones 32x16 (layout-invariant), so D = row sums of A.
  v16h bOnes;
#pragma unroll
  for (int i = 0; i < 16; ++i) bOnes[i] = (_Float16)1.f;

  const _Float16* wBase = wA + (long)(cobase + mrow) * KK + klo;

  for (int cb = 0; cb < 5; ++cb) {        // 5 blocks of 4 columns
    const int col0 = col00 + cb * 4;
    const _Float16* pRow = &P[col0 * KK + klo];
    v8f a0 = {}, a1 = {}, a2 = {}, a3 = {};
    for (int c = 0; c < NCHK; ++c) {
      const _Float16* wp = wBase + c * 32;
      const v8h w0 = *(const v8h*)wp;          // k runs klo+{0..7}
      const v8h w1 = *(const v8h*)(wp + 16);   // and klo+16+{0..7}
      if (c + 1 < NCHK) __builtin_prefetch(wp + 32, 0, 3);

      // Phase 1: all LDS loads for the 4 pixels (broadcast ds_load_b128).
      const _Float16* pc = pRow + c * 32;
      const v8h q00 = *(const v8h*)(pc);
      const v8h q01 = *(const v8h*)(pc + 16);
      const v8h q10 = *(const v8h*)(pc + KK);
      const v8h q11 = *(const v8h*)(pc + KK + 16);
      const v8h q20 = *(const v8h*)(pc + 2 * KK);
      const v8h q21 = *(const v8h*)(pc + 2 * KK + 16);
      const v8h q30 = *(const v8h*)(pc + 3 * KK);
      const v8h q31 = *(const v8h*)(pc + 3 * KK + 16);

      // Phase 2: four *distinct live* A matrices -> disjoint register sets,
      // so no WMMA->VALU WAR hazard NOPs when A is rebuilt next chunk.
      const v16h A0 = mkAbsA(q00, q01, w0, w1);
      const v16h A1 = mkAbsA(q10, q11, w0, w1);
      const v16h A2 = mkAbsA(q20, q21, w0, w1);
      const v16h A3 = mkAbsA(q30, q31, w0, w1);

      // Phase 3: four independent-accumulator WMMAs back-to-back.
      a0 = __builtin_amdgcn_wmma_f32_16x16x32_f16(false, A0, false, bOnes,
                                                  (short)0, a0, false, false);
      a1 = __builtin_amdgcn_wmma_f32_16x16x32_f16(false, A1, false, bOnes,
                                                  (short)0, a1, false, false);
      a2 = __builtin_amdgcn_wmma_f32_16x16x32_f16(false, A2, false, bOnes,
                                                  (short)0, a2, false, false);
      a3 = __builtin_amdgcn_wmma_f32_16x16x32_f16(false, A3, false, bOnes,
                                                  (short)0, a3, false, false);
    }
    // All D columns are identical (B = ones). Lane 0 holds rows m=0..7 in
    // acc[0..7]; lane 16 holds rows m=8..15.
    if (lane == 0 || lane == 16) {
      const int mofs = (lane == 16) ? 8 : 0;
#pragma unroll
      for (int px = 0; px < 4; ++px) {
        const v8f acc = px == 0 ? a0 : px == 1 ? a1 : px == 2 ? a2 : a3;
        const int col = col0 + px;
#pragma unroll
        for (int v = 0; v < 8; ++v) {
          const int co = cobase + mofs + v;
          const long o = (((long)(n * CC + co)) * HH + row) * WW + col;
          const float rel = fmaxf(-acc[v], 0.f);  // relu(-SAD)
          if constexpr (FINAL) {
            ((float*)outRaw)[o] = 0.1f * rel + xres[o];
          } else {
            ((_Float16*)outRaw)[o] = (_Float16)rel;
          }
        }
      }
    }
  }
}

// ---------------------------------------------------------------------------
// Launch: prep f16 weights, layer1 (x,w1 -> f16 y1 in ws), layer2 fused.
// ws layout: [0,72KB) wA1 | [72KB,144KB) wA2 | [144KB, +800KB) y1 (f16)
// ---------------------------------------------------------------------------
extern "C" void kernel_launch(void* const* d_in, const int* in_sizes, int n_in,
                              void* d_out, int out_size, void* d_ws,
                              size_t ws_size, hipStream_t stream) {
  (void)in_sizes; (void)n_in; (void)out_size; (void)ws_size;
  const float* x  = (const float*)d_in[0];
  const float* w1 = (const float*)d_in[1];
  const float* w2 = (const float*)d_in[2];

  char* ws = (char*)d_ws;
  _Float16* wA1 = (_Float16*)ws;                     // 64*576*2 = 73728 B
  _Float16* wA2 = (_Float16*)(ws + 73728);
  _Float16* y1  = (_Float16*)(ws + 147456);          // 4*64*40*40*2 B

  const int nw = CC * KK;                            // 36864
  cvt_weights<<<(nw + 255) / 256, 256, 0, stream>>>(w1, wA1, nw);
  cvt_weights<<<(nw + 255) / 256, 256, 0, stream>>>(w2, wA2, nw);

  adder_layer<false, false><<<NB * HH, 256, 0, stream>>>(x, nullptr, wA1, y1);
  adder_layer<true,  true ><<<NB * HH, 256, 0, stream>>>(y1, x, wA2, d_out);
}
